// RNNTBeamSearch_49366354100329
// MI455X (gfx1250) — matrix operand order repaired and checked by reading, not compiled, not run on hardware
//
#include <hip/hip_runtime.h>
#include <hip/hip_bf16.h>
#include <math.h>

// ---------------------------------------------------------------------------
// RNN-T beam search for MI455X (gfx1250).
// Matmuls (joint: 16x1024x8192, LSTM gates: 16x2048x4096, encoder:
// 512x512x1024) run on v_wmma_f32_16x16x32_f16 with f16 weights pre-packed
// into WMMA B-panel layout (kept resident in the 192MB L2 across all 512
// steps). Beam state (h, c, pred_out, scores, tokens) ping-pongs in f32.
// ---------------------------------------------------------------------------

#define TT    512
#define DD    512
#define HH    1024
#define VV    8192
#define BB    16
#define BLANK (VV - 1)
#define NEGV  (-99999.0f)
#define EXPB  (10.0f)
#define INVTEMP (1.0f)

typedef __attribute__((ext_vector_type(8)))  _Float16 v8h;
typedef __attribute__((ext_vector_type(16))) _Float16 v16h;
typedef __attribute__((ext_vector_type(8)))  float    v8f;

__device__ __forceinline__ float sigmoidf_(float x) { return 1.0f / (1.0f + __expf(-x)); }

// ---------------------------------------------------------------------------
// Generic WMMA GEMM:  C[M][N] = A[M][K] (f16 row-major) * Bpanel + bias
// One wave per 16x16 output tile; K consumed in 32-chunks via
// v_wmma_f32_16x16x32_f16.
//
// B-panel layout: tile (nt, kt) occupies 32 lanes x 16 halfs contiguous:
//   element j of lane l is  W[kt*32 + (j>>3)*16 + (l>>4)*8 + (j&7)][nt*16 + (l&15)]
// A-fragment layout (ISA 16-bit A 16x32): lane l holds row (l&15),
//   K = kt*32 + (l>>4)*8 + j     for j=0..7
//   K = kt*32 + 16 + (l>>4)*8 + (j-8) for j=8..15
// ---------------------------------------------------------------------------
__global__ void wmma_gemm_kernel(const _Float16* __restrict__ A,
                                 const _Float16* __restrict__ Bp,
                                 const float* __restrict__ bias,
                                 float* __restrict__ C,
                                 int M, int N, int K)
{
    const int wave = (blockIdx.x * blockDim.x + threadIdx.x) >> 5;
    const int lane = threadIdx.x & 31;
    const int ntiles = N >> 4;
    const int ktiles = K >> 5;
    const int total  = (M >> 4) * ntiles;
    if (wave >= total) return;                 // wave-uniform: EXEC all-ones inside
    const int mt = wave / ntiles;
    const int nt = wave % ntiles;

    v8f acc = {};
    const _Float16* __restrict__ Abase =
        A + (size_t)(mt * 16 + (lane & 15)) * K + ((lane >> 4) << 3);
    const _Float16* __restrict__ Bbase =
        Bp + ((size_t)nt * ktiles * 32 + lane) * 16;

    for (int kt = 0; kt < ktiles; ++kt) {
        const _Float16* ap = Abase + kt * 32;
        v8h a0 = *(const v8h*)(ap);
        v8h a1 = *(const v8h*)(ap + 16);
        v16h a;
#pragma unroll
        for (int j = 0; j < 8; ++j) { a[j] = a0[j]; a[8 + j] = a1[j]; }
        v16h b = *(const v16h*)(Bbase + (size_t)kt * 32 * 16);
        acc = __builtin_amdgcn_wmma_f32_16x16x32_f16(
            false, a, false, b, (short)0, acc, false, false);
    }

    const int n    = nt * 16 + (lane & 15);
    const int mrow = mt * 16 + ((lane >> 4) << 3);
    const float bv = bias ? bias[n] : 0.0f;
#pragma unroll
    for (int r = 0; r < 8; ++r)
        C[(size_t)(mrow + r) * N + n] = acc[r] + bv;
}

// ---------------------------------------------------------------------------
// Pack f32 weight matrix (rows 0..K1-1 from W1, rows K1..K-1 from W2, both
// row-major with N columns) into the f16 B-panel layout above.
// ---------------------------------------------------------------------------
__global__ void pack_b_kernel(const float* __restrict__ W1,
                              const float* __restrict__ W2,
                              _Float16* __restrict__ Bp,
                              int K1, int K, int N)
{
    size_t idx = (size_t)blockIdx.x * blockDim.x + threadIdx.x;
    size_t total = (size_t)K * N;
    if (idx >= total) return;
    int j    = (int)(idx & 15);
    int lane = (int)((idx >> 4) & 31);
    size_t tile = idx >> 9;
    int ktiles = K >> 5;
    int kt = (int)(tile % ktiles);
    int nt = (int)(tile / ktiles);
    int n = nt * 16 + (lane & 15);
    int k = kt * 32 + ((j >> 3) << 4) + ((lane >> 4) << 3) + (j & 7);
    float v = (k < K1) ? W1[(size_t)k * N + n] : W2[(size_t)(k - K1) * N + n];
    Bp[idx] = (_Float16)v;
}

__global__ void cvt_f16_kernel(const float* __restrict__ src,
                               _Float16* __restrict__ dst, size_t n)
{
    size_t i = (size_t)blockIdx.x * blockDim.x + threadIdx.x;
    if (i < n) dst[i] = (_Float16)src[i];
}

// A for joint: relu(enc[t] + pred_out)  -> f16 [16][1024]
__global__ void prep_joint_kernel(const float* __restrict__ enc, int t,
                                  const float* __restrict__ pred,
                                  _Float16* __restrict__ Aj)
{
    int i = blockIdx.x * blockDim.x + threadIdx.x;
    if (i >= BB * HH) return;
    int k = i & (HH - 1);
    float v = enc[(size_t)t * HH + k] + pred[i];
    Aj[i] = (_Float16)fmaxf(v, 0.0f);
}

// Per-beam softmax statistics + blank scores + expand-beam threshold.
__global__ void softmax_stats_kernel(const float* __restrict__ logits,
                                     const float* __restrict__ scores,
                                     float* __restrict__ denom,
                                     float* __restrict__ thr,
                                     float* __restrict__ blank_sc)
{
    __shared__ float rmax[256], rnb[256];
    const int b = blockIdx.x, tid = threadIdx.x;
    const float* row = logits + (size_t)b * VV;
    float mx = -3.0e38f, mnb = -3.0e38f;
    for (int v = tid; v < VV; v += 256) {
        float x = row[v] * INVTEMP;
        mx = fmaxf(mx, x);
        if (v >= 4 && v < VV - 1) mnb = fmaxf(mnb, x);
    }
    rmax[tid] = mx; rnb[tid] = mnb;
    __syncthreads();
    for (int s = 128; s > 0; s >>= 1) {
        if (tid < s) {
            rmax[tid] = fmaxf(rmax[tid], rmax[tid + s]);
            rnb[tid]  = fmaxf(rnb[tid],  rnb[tid + s]);
        }
        __syncthreads();
    }
    mx = rmax[0]; mnb = rnb[0];
    __syncthreads();
    float s = 0.0f;
    for (int v = tid; v < VV; v += 256) s += __expf(row[v] * INVTEMP - mx);
    rmax[tid] = s;
    __syncthreads();
    for (int st = 128; st > 0; st >>= 1) {
        if (tid < st) rmax[tid] += rmax[tid + st];
        __syncthreads();
    }
    if (tid == 0) {
        float dn = mx + __logf(rmax[0]);
        denom[b]    = dn;
        thr[b]      = mnb - dn - EXPB;
        blank_sc[b] = scores[b] + (row[VV - 1] * INVTEMP - dn);
    }
}

// Candidate array: cand[b*(V-1)+v] = scores[b] + pruned logp
__global__ void build_cand_kernel(const float* __restrict__ logits,
                                  const float* __restrict__ denom,
                                  const float* __restrict__ thr,
                                  const float* __restrict__ scores,
                                  float* __restrict__ cand)
{
    int i = blockIdx.x * blockDim.x + threadIdx.x;
    const int NC = BB * (VV - 1);
    if (i >= NC) return;
    int b = i / (VV - 1), v = i % (VV - 1);
    float lp = logits[(size_t)b * VV + v] * INVTEMP - denom[b];
    if (v < 4) lp += NEGV;
    float val = (lp > thr[b]) ? lp : NEGV;
    cand[i] = scores[b] + val;
}

// Iterative block-wide argmax top-16 over 16*8191 candidates (first-index ties).
__global__ void topk_cand_kernel(float* __restrict__ cand,
                                 float* __restrict__ nb_scores,
                                 int* __restrict__ nb_tok,
                                 int* __restrict__ nb_hypo)
{
    __shared__ float sv[1024];
    __shared__ int   si[1024];
    const int NC = BB * (VV - 1);
    const int tid = threadIdx.x;
    for (int r = 0; r < BB; ++r) {
        float bv = -3.0e38f; int bi = 0x7fffffff;
        for (int i = tid; i < NC; i += 1024) {
            float v = cand[i];
            if (v > bv) { bv = v; bi = i; }
        }
        sv[tid] = bv; si[tid] = bi;
        __syncthreads();
        for (int s = 512; s > 0; s >>= 1) {
            if (tid < s) {
                if (sv[tid + s] > sv[tid] ||
                    (sv[tid + s] == sv[tid] && si[tid + s] < si[tid])) {
                    sv[tid] = sv[tid + s]; si[tid] = si[tid + s];
                }
            }
            __syncthreads();
        }
        if (tid == 0) {
            int idx = si[0];
            nb_scores[r] = sv[0];
            nb_hypo[r]   = idx / (VV - 1);
            nb_tok[r]    = idx % (VV - 1);
            cand[idx]    = -3.0e38f;
        }
        __syncthreads();
    }
}

// A for LSTM gates: [ embed_f16[nb_tok[m]] | (f16)h[nb_hypo[m]] ]  (16 x 2048)
__global__ void prep_lstm_kernel(const _Float16* __restrict__ embed_h,
                                 const float* __restrict__ h_cur,
                                 const int* __restrict__ nb_tok,
                                 const int* __restrict__ nb_hypo,
                                 _Float16* __restrict__ Al)
{
    int i = blockIdx.x * blockDim.x + threadIdx.x;
    if (i >= BB * 2 * HH) return;
    int m = i >> 11, k = i & (2 * HH - 1);
    _Float16 v;
    if (k < HH) v = embed_h[(size_t)nb_tok[m] * HH + k];
    else        v = (_Float16)h_cur[(size_t)nb_hypo[m] * HH + (k - HH)];
    Al[i] = v;
}

__global__ void lstm_act_kernel(const float* __restrict__ gates,
                                const float* __restrict__ c_cur,
                                const int* __restrict__ nb_hypo,
                                float* __restrict__ new_h,
                                float* __restrict__ new_c)
{
    int i = blockIdx.x * blockDim.x + threadIdx.x;
    if (i >= BB * HH) return;
    int m = i >> 10, k = i & (HH - 1);
    const float* g = gates + (size_t)m * 4 * HH;
    float gi = g[k], gf = g[HH + k], gg = g[2 * HH + k], go = g[3 * HH + k];
    float cp = c_cur[(size_t)nb_hypo[m] * HH + k];
    float c2 = sigmoidf_(gf) * cp + sigmoidf_(gi) * tanhf(gg);
    new_c[i] = c2;
    new_h[i] = sigmoidf_(go) * tanhf(c2);
}

// Merge blank/non-blank hypotheses (32 -> 16) and gather next-state buffers.
__global__ void select_kernel(int t,
                              const float* __restrict__ blank_sc,
                              const float* __restrict__ nb_scores,
                              const int* __restrict__ nb_tok,
                              const int* __restrict__ nb_hypo,
                              const int* __restrict__ tok_cur, int* __restrict__ tok_nxt,
                              const float* __restrict__ pred_cur, float* __restrict__ pred_nxt,
                              const float* __restrict__ h_cur, float* __restrict__ h_nxt,
                              const float* __restrict__ c_cur, float* __restrict__ c_nxt,
                              const float* __restrict__ new_h,
                              const float* __restrict__ new_c,
                              float* __restrict__ scores_nxt)
{
    __shared__ int s_srctok[BB], s_settok[BB], s_isb[BB], s_bsel[BB], s_nsel[BB];
    if (threadIdx.x == 0) {
        float all[2 * BB]; int used[2 * BB];
        for (int i = 0; i < BB; ++i) { all[i] = blank_sc[i]; all[BB + i] = nb_scores[i]; }
        for (int i = 0; i < 2 * BB; ++i) used[i] = 0;
        for (int r = 0; r < BB; ++r) {
            float bv = -3.0e38f; int bi = 0;
            for (int i = 0; i < 2 * BB; ++i)
                if (!used[i] && all[i] > bv) { bv = all[i]; bi = i; }
            used[bi] = 1;
            scores_nxt[r] = bv;
            int isb = (bi < BB);
            int bs  = isb ? bi : (BB - 1);
            int ns  = isb ? 0 : (bi - BB);
            s_isb[r] = isb; s_bsel[r] = bs; s_nsel[r] = ns;
            s_srctok[r] = isb ? bs : nb_hypo[ns];
            s_settok[r] = isb ? -1 : nb_tok[ns];
        }
    }
    __syncthreads();
    for (int i = threadIdx.x; i < BB * TT; i += blockDim.x) {
        int row = i / TT, col = i % TT;
        int val = tok_cur[s_srctok[row] * TT + col];
        if (col == t && s_settok[row] >= 0) val = s_settok[row];
        tok_nxt[i] = val;
    }
    for (int i = threadIdx.x; i < BB * HH; i += blockDim.x) {
        int row = i / HH, col = i % HH;
        int isb = s_isb[row], bs = s_bsel[row], ns = s_nsel[row];
        pred_nxt[i] = isb ? pred_cur[(size_t)bs * HH + col] : new_h[(size_t)ns * HH + col];
        h_nxt[i]    = isb ? h_cur[(size_t)bs * HH + col]    : new_h[(size_t)ns * HH + col];
        c_nxt[i]    = isb ? c_cur[(size_t)bs * HH + col]    : new_c[(size_t)ns * HH + col];
    }
}

__global__ void init_pre_kernel(float* __restrict__ h, float* __restrict__ c,
                                int* __restrict__ nb_tok, int* __restrict__ nb_hypo)
{
    int i = blockIdx.x * blockDim.x + threadIdx.x;
    if (i < BB * HH) { h[i] = 0.0f; c[i] = 0.0f; }
    if (i < BB)      { nb_tok[i] = BLANK; nb_hypo[i] = i; }
}

__global__ void init_post_kernel(const float* __restrict__ new_h,
                                 const float* __restrict__ new_c,
                                 float* __restrict__ pred, float* __restrict__ h,
                                 float* __restrict__ c, float* __restrict__ scores,
                                 int* __restrict__ tokens)
{
    int i = blockIdx.x * blockDim.x + threadIdx.x;
    if (i < BB * HH) { pred[i] = new_h[i]; h[i] = new_h[i]; c[i] = new_c[i]; }
    if (i < BB)      scores[i] = (i == 0) ? 0.0f : -1.0e9f;
    if (i < BB * TT) tokens[i] = BLANK;
}

__global__ void finalize_kernel(const float* __restrict__ scores,
                                const int* __restrict__ tokens,
                                float* __restrict__ out)
{
    int i = blockIdx.x * blockDim.x + threadIdx.x;
    if (i < BB) out[i] = scores[i];
    if (i < BB * TT) ((int*)(out + BB))[i] = tokens[i];
}

// ---------------------------------------------------------------------------
extern "C" void kernel_launch(void* const* d_in, const int* in_sizes, int n_in,
                              void* d_out, int out_size, void* d_ws, size_t ws_size,
                              hipStream_t stream)
{
    (void)in_sizes; (void)n_in; (void)out_size; (void)ws_size;
    const float* input   = (const float*)d_in[0];
    const float* enc_W   = (const float*)d_in[1];
    const float* enc_b   = (const float*)d_in[2];
    const float* embed   = (const float*)d_in[3];
    const float* Wx      = (const float*)d_in[4];
    const float* Wh      = (const float*)d_in[5];
    const float* b_lstm  = (const float*)d_in[6];
    const float* joint_W = (const float*)d_in[7];
    const float* joint_b = (const float*)d_in[8];
    float* out = (float*)d_out;

    char* w = (char*)d_ws;
    size_t off = 0;
    auto take = [&](size_t bytes) -> char* {
        char* p = w + off;
        off = (off + bytes + 255) & ~(size_t)255;
        return p;
    };
    _Float16* jointW_p = (_Float16*)take((size_t)HH * VV * 2);
    _Float16* wxwh_p   = (_Float16*)take((size_t)2 * HH * 4 * HH * 2);
    _Float16* encW_p   = (_Float16*)take((size_t)DD * HH * 2);
    _Float16* embed_h  = (_Float16*)take((size_t)VV * HH * 2);
    _Float16* inA      = (_Float16*)take((size_t)TT * DD * 2);
    float* enc    = (float*)take((size_t)TT * HH * 4);
    _Float16* Aj  = (_Float16*)take((size_t)BB * HH * 2);
    _Float16* Al  = (_Float16*)take((size_t)BB * 2 * HH * 2);
    float* logits = (float*)take((size_t)BB * VV * 4);
    float* gates  = (float*)take((size_t)BB * 4 * HH * 4);
    float* cand   = (float*)take((size_t)BB * (VV - 1) * 4);
    float* denom  = (float*)take(BB * 4);
    float* thr    = (float*)take(BB * 4);
    float* blank  = (float*)take(BB * 4);
    float* nbsc   = (float*)take(BB * 4);
    int*   nbtok  = (int*)take(BB * 4);
    int*   nbhyp  = (int*)take(BB * 4);
    float* new_h  = (float*)take((size_t)BB * HH * 4);
    float* new_c  = (float*)take((size_t)BB * HH * 4);
    float* hbuf[2]; float* cbuf[2]; float* pbuf[2]; float* sbuf[2]; int* tbuf[2];
    for (int i = 0; i < 2; ++i) {
        hbuf[i] = (float*)take((size_t)BB * HH * 4);
        cbuf[i] = (float*)take((size_t)BB * HH * 4);
        pbuf[i] = (float*)take((size_t)BB * HH * 4);
        sbuf[i] = (float*)take(BB * 4);
        tbuf[i] = (int*)take((size_t)BB * TT * 4);
    }

    auto gemm = [&](const _Float16* A, const _Float16* Bp, const float* bias,
                    float* C, int M, int N, int K) {
        int tiles = (M / 16) * (N / 16);
        int blocks = (tiles + 7) / 8;                 // 8 waves per 256-thread block
        wmma_gemm_kernel<<<blocks, 256, 0, stream>>>(A, Bp, bias, C, M, N, K);
    };

    // ---- one-time prep: f16 conversions + B-panel packing -------------------
    {
        size_t n1 = (size_t)TT * DD;
        cvt_f16_kernel<<<(int)((n1 + 255) / 256), 256, 0, stream>>>(input, inA, n1);
        size_t n2 = (size_t)VV * HH;
        cvt_f16_kernel<<<(int)((n2 + 255) / 256), 256, 0, stream>>>(embed, embed_h, n2);
        size_t ne = (size_t)DD * HH;
        pack_b_kernel<<<(int)((ne + 255) / 256), 256, 0, stream>>>(enc_W, enc_W, encW_p, DD, DD, HH);
        size_t nj = (size_t)HH * VV;
        pack_b_kernel<<<(int)((nj + 255) / 256), 256, 0, stream>>>(joint_W, joint_W, jointW_p, HH, HH, VV);
        size_t nl = (size_t)2 * HH * 4 * HH;
        pack_b_kernel<<<(int)((nl + 255) / 256), 256, 0, stream>>>(Wx, Wh, wxwh_p, HH, 2 * HH, 4 * HH);
    }

    // ---- encoder: enc = input @ enc_W + enc_b  (512x512x1024) ---------------
    gemm(inA, encW_p, enc_b, enc, TT, HH, DD);

    // ---- predictor init: p0, pc0 = LSTM(embed[BLANK], 0, 0) -----------------
    init_pre_kernel<<<(BB * HH + 255) / 256, 256, 0, stream>>>(hbuf[0], cbuf[0], nbtok, nbhyp);
    prep_lstm_kernel<<<(BB * 2 * HH + 255) / 256, 256, 0, stream>>>(embed_h, hbuf[0], nbtok, nbhyp, Al);
    gemm(Al, wxwh_p, b_lstm, gates, BB, 4 * HH, 2 * HH);
    lstm_act_kernel<<<(BB * HH + 255) / 256, 256, 0, stream>>>(gates, cbuf[0], nbhyp, new_h, new_c);
    init_post_kernel<<<(BB * TT + 255) / 256, 256, 0, stream>>>(new_h, new_c, pbuf[0], hbuf[0],
                                                                cbuf[0], sbuf[0], tbuf[0]);

    // ---- 512 sequential decode steps ---------------------------------------
    for (int t = 0; t < TT; ++t) {
        int cur = t & 1, nxt = cur ^ 1;
        prep_joint_kernel<<<(BB * HH + 255) / 256, 256, 0, stream>>>(enc, t, pbuf[cur], Aj);
        gemm(Aj, jointW_p, joint_b, logits, BB, VV, HH);
        softmax_stats_kernel<<<BB, 256, 0, stream>>>(logits, sbuf[cur], denom, thr, blank);
        build_cand_kernel<<<(BB * (VV - 1) + 255) / 256, 256, 0, stream>>>(logits, denom, thr,
                                                                           sbuf[cur], cand);
        topk_cand_kernel<<<1, 1024, 0, stream>>>(cand, nbsc, nbtok, nbhyp);
        prep_lstm_kernel<<<(BB * 2 * HH + 255) / 256, 256, 0, stream>>>(embed_h, hbuf[cur],
                                                                        nbtok, nbhyp, Al);
        gemm(Al, wxwh_p, b_lstm, gates, BB, 4 * HH, 2 * HH);
        lstm_act_kernel<<<(BB * HH + 255) / 256, 256, 0, stream>>>(gates, cbuf[cur], nbhyp,
                                                                   new_h, new_c);
        select_kernel<<<1, 256, 0, stream>>>(t, blank, nbsc, nbtok, nbhyp,
                                             tbuf[cur], tbuf[nxt],
                                             pbuf[cur], pbuf[nxt],
                                             hbuf[cur], hbuf[nxt],
                                             cbuf[cur], cbuf[nxt],
                                             new_h, new_c, sbuf[nxt]);
    }

    // after t = 511 (odd), results land in buffer index 0
    finalize_kernel<<<(BB * TT + 255) / 256, 256, 0, stream>>>(sbuf[0], tbuf[0], out);
}
